// SetEncoder_44882408243510
// MI455X (gfx1250) — matrix operand order repaired and checked by reading, not compiled, not run on hardware
//
#include <hip/hip_runtime.h>

// ---------------------------------------------------------------------------
// SetEncoder on MI455X (gfx1250, wave32)
//
// Factorization: pooled = (sum_p m_p * relu(feats_p @ W1 + b1)) @ W2 * sA + b2 * sB
//   where sA = 1/max(cnt,1), sB = (cnt>0 ? 1 : 0).
// Kernel 1: feature build + 6->64 MLP + masked pool (VALU/TRANS; K=6 too small
//           for WMMA). Writes pooled hidden (f16) + scales into d_ws.
// Kernel 2: 65536x64 @ 64x128 GEMM via v_wmma_f32_16x16x32_f16.
//           A tile staged HBM->LDS with global_load_async_to_lds_b128
//           (ASYNCcnt path), W2 pre-swizzled into WMMA B-fragment order so
//           every fragment load is a ds_load_b128.
// ---------------------------------------------------------------------------

#define NSETS   65536
#define NPLY    22
#define NIN     6
#define NHID    64
#define NOUT    128

typedef __attribute__((ext_vector_type(16))) _Float16 v16h;
typedef __attribute__((ext_vector_type(8)))  _Float16 v8h;
typedef __attribute__((ext_vector_type(8)))  float    v8f;

union V16U { v16h v; v8h h[2]; _Float16 e[16]; };

// ---------------------------------------------------------------------------
// Kernel 1: one thread per set. Pools relu(hidden) over valid players.
// ---------------------------------------------------------------------------
__global__ __launch_bounds__(256) void se_feat_pool_kernel(
    const float* __restrict__ player_locs,  // [B,P,2]
    const float* __restrict__ actor_locs,   // [B,2]
    const float* __restrict__ flags,        // [B,P,2]
    const int*   __restrict__ mask,         // [B,P]
    const float* __restrict__ W1,           // [6,64]
    const float* __restrict__ b1,           // [64]
    _Float16*    __restrict__ Hp,           // [B,64]  (workspace)
    float*       __restrict__ scales)       // [B,2]   (workspace)
{
    __shared__ float sW1[NIN * NHID];
    __shared__ float sb1[NHID];

    for (int i = threadIdx.x; i < NIN * NHID; i += blockDim.x) sW1[i] = W1[i];
    for (int i = threadIdx.x; i < NHID;       i += blockDim.x) sb1[i] = b1[i];
    __syncthreads();

    const int b = blockIdx.x * blockDim.x + threadIdx.x;
    if (b >= NSETS) return;

    const float ax = actor_locs[2 * b + 0];
    const float ay = actor_locs[2 * b + 1];

    float pool[NHID];
#pragma unroll
    for (int j = 0; j < NHID; ++j) pool[j] = 0.0f;
    float cnt = 0.0f;

    for (int p = 0; p < NPLY; ++p) {
        const int   idx = b * NPLY + p;
        const float m   = (float)mask[idx];
        if (m == 0.0f) continue;
        cnt += 1.0f;

        const float dx   = player_locs[2 * idx + 0] - ax;
        const float dy   = player_locs[2 * idx + 1] - ay;
        const float dist = sqrtf(dx * dx + dy * dy);
        const float ang  = atan2f(dy, dx);
        const float f4   = flags[2 * idx + 0];
        const float f5   = flags[2 * idx + 1];

        // h = b1 + feats @ W1 ; pool += relu(h).  float4 LDS reads -> ds_load_b128.
#pragma unroll
        for (int j = 0; j < NHID; j += 4) {
            const float4 w0 = *(const float4*)&sW1[0 * NHID + j];
            const float4 w1 = *(const float4*)&sW1[1 * NHID + j];
            const float4 w2 = *(const float4*)&sW1[2 * NHID + j];
            const float4 w3 = *(const float4*)&sW1[3 * NHID + j];
            const float4 w4 = *(const float4*)&sW1[4 * NHID + j];
            const float4 w5 = *(const float4*)&sW1[5 * NHID + j];
            const float4 bb = *(const float4*)&sb1[j];
            float h0 = bb.x + dx*w0.x + dy*w1.x + dist*w2.x + ang*w3.x + f4*w4.x + f5*w5.x;
            float h1 = bb.y + dx*w0.y + dy*w1.y + dist*w2.y + ang*w3.y + f4*w4.y + f5*w5.y;
            float h2 = bb.z + dx*w0.z + dy*w1.z + dist*w2.z + ang*w3.z + f4*w4.z + f5*w5.z;
            float h3 = bb.w + dx*w0.w + dy*w1.w + dist*w2.w + ang*w3.w + f4*w4.w + f5*w5.w;
            pool[j + 0] += fmaxf(h0, 0.0f);
            pool[j + 1] += fmaxf(h1, 0.0f);
            pool[j + 2] += fmaxf(h2, 0.0f);
            pool[j + 3] += fmaxf(h3, 0.0f);
        }
    }

#pragma unroll
    for (int j = 0; j < NHID; ++j) Hp[b * NHID + j] = (_Float16)pool[j];
    scales[2 * b + 0] = 1.0f / fmaxf(cnt, 1.0f);     // GEMM-result scale
    scales[2 * b + 1] = (cnt > 0.0f) ? 1.0f : 0.0f;  // b2 scale
}

// ---------------------------------------------------------------------------
// Kernel 2: [B,64]f16 @ [64,128]f16 -> [B,128]f32 with v_wmma_f32_16x16x32_f16.
// 8 waves/block, each wave owns a 16-row M-tile x all 128 cols:
//   8 N-tiles x 2 K-steps = 16 WMMA per wave.
// All fragment loads are 16-byte ds_load_b128 thanks to the swizzled layouts.
// ---------------------------------------------------------------------------
__global__ __launch_bounds__(256) void se_gemm2_wmma_kernel(
    const _Float16* __restrict__ Hp,      // [B,64]
    const float*    __restrict__ scales,  // [B,2]
    const float*    __restrict__ W2,      // [64,128]
    const float*    __restrict__ b2,      // [128]
    float*          __restrict__ out)     // [B,128]
{
    __shared__ _Float16 sA[128 * NHID];    // 128 rows of pooled hidden (16 KB)
    // W2 pre-swizzled into B-fragment order: [nt][ks][lane][e] (16 KB)
    __shared__ _Float16 sBf[8 * 2 * 32 * 16];
    __shared__ float    sb2[NOUT];

    const int tid      = threadIdx.x;
    const int rowBlock = blockIdx.x * 128;

    // ---- Stage A tile via async HBM->LDS DMA (no VGPR transit, ASYNCcnt). ----
    {
        const unsigned long long srcBase =
            (unsigned long long)(uintptr_t)(Hp + (size_t)rowBlock * NHID);
        const unsigned ldsBase = (unsigned)(uintptr_t)(&sA[0]);
        // 128 rows * 64 halves * 2B = 16384 B = 1024 chunks of 16 B.
#pragma unroll
        for (int c = 0; c < 4; ++c) {
            const int i = tid + c * 256;
            const unsigned goff = (unsigned)i * 16u;
            const unsigned loff = ldsBase + (unsigned)i * 16u;
            asm volatile("global_load_async_to_lds_b128 %0, %1, %2"
                         :: "v"(loff), "v"(goff), "s"(srcBase) : "memory");
        }
    }

    // ---- Stage W2 (f32 -> f16) scattered into WMMA B-fragment order. ----
    // Fragment (nt, ks): element e of lane (kg*16 + r) = W2[ks*32+kg*16+e][nt*16+r]
    for (int i = tid; i < NHID * NOUT; i += 256) {
        const int k  = i >> 7;       // 0..63
        const int n  = i & 127;
        const int nt = n >> 4;
        const int r  = n & 15;
        const int ks = k >> 5;
        const int kg = (k >> 4) & 1;
        const int e  = k & 15;
        sBf[(((nt << 1) | ks) * 32 + kg * 16 + r) * 16 + e] = (_Float16)W2[i];
    }
    for (int i = tid; i < NOUT; i += 256) sb2[i] = b2[i];

    // Hint the next block's A tile toward the caches (global_prefetch_b8).
    if ((int)blockIdx.x + 1 < (int)gridDim.x)
        __builtin_prefetch(Hp + (rowBlock + 128) * NHID, 0, 1);

    // Fence the async DMA, then rendezvous all 8 waves.
    asm volatile("s_wait_asynccnt 0" ::: "memory");
    __syncthreads();

    const int wave   = tid >> 5;        // 0..7
    const int lane   = tid & 31;
    const int kg     = lane >> 4;       // lane group 0/1
    const int r      = lane & 15;       // N index within tile
    const int rlocal = wave * 16 + r;   // A row within this block's 128-row tile

    // A fragments (contiguous K order, matching B): per lane two contiguous 8-half
    // runs -> 2 x ds_load_b128 each.
    //  lanes 0-15 : K = 0..15  (of the K-step's 32)
    //  lanes 16-31: K = 16..31
    V16U a0, a1;
    a0.h[0] = *(const v8h*)&sA[rlocal * NHID +      kg * 16];
    a0.h[1] = *(const v8h*)&sA[rlocal * NHID +  8 + kg * 16];
    a1.h[0] = *(const v8h*)&sA[rlocal * NHID + 32 + kg * 16];
    a1.h[1] = *(const v8h*)&sA[rlocal * NHID + 40 + kg * 16];

    // Per-row epilogue scales: C/D VGPR j holds row M = j + kg*8.
    float sAcc[8], sBias[8];
#pragma unroll
    for (int j = 0; j < 8; ++j) {
        const int row = rowBlock + wave * 16 + kg * 8 + j;
        sAcc[j]  = scales[2 * row + 0];
        sBias[j] = scales[2 * row + 1];
    }

#pragma unroll
    for (int nt = 0; nt < 8; ++nt) {
        // B fragments: contiguous per lane in the swizzled layout -> b128 loads.
        V16U bf0, bf1;
        {
            const int base0 = (((nt << 1) | 0) * 32 + lane) * 16;
            const int base1 = (((nt << 1) | 1) * 32 + lane) * 16;
            bf0.h[0] = *(const v8h*)&sBf[base0];
            bf0.h[1] = *(const v8h*)&sBf[base0 + 8];
            bf1.h[0] = *(const v8h*)&sBf[base1];
            bf1.h[1] = *(const v8h*)&sBf[base1 + 8];
        }

        v8f c = {0.f, 0.f, 0.f, 0.f, 0.f, 0.f, 0.f, 0.f};
        c = __builtin_amdgcn_wmma_f32_16x16x32_f16(
                false, a0.v, false, bf0.v, (short)0, c, false, false);
        c = __builtin_amdgcn_wmma_f32_16x16x32_f16(
                false, a1.v, false, bf1.v, (short)0, c, false, false);

#pragma unroll
        for (int j = 0; j < 8; ++j) {
            const int row = rowBlock + wave * 16 + kg * 8 + j;
            out[row * NOUT + nt * 16 + r] =
                c[j] * sAcc[j] + sb2[nt * 16 + r] * sBias[j];
        }
    }
}

extern "C" void kernel_launch(void* const* d_in, const int* in_sizes, int n_in,
                              void* d_out, int out_size, void* d_ws, size_t ws_size,
                              hipStream_t stream) {
    const float* player_locs = (const float*)d_in[0];
    const float* actor_locs  = (const float*)d_in[1];
    const float* flags       = (const float*)d_in[2];
    const int*   mask        = (const int*)  d_in[3];
    const float* W1          = (const float*)d_in[4];
    const float* b1          = (const float*)d_in[5];
    const float* W2          = (const float*)d_in[6];
    const float* b2          = (const float*)d_in[7];
    float*       out         = (float*)d_out;

    // Workspace layout: Hp [B,64] f16, then scales [B,2] f32.
    _Float16* Hp     = (_Float16*)d_ws;
    float*    scales = (float*)((char*)d_ws + (size_t)NSETS * NHID * sizeof(_Float16));

    // Kernel 1: one thread per set.
    se_feat_pool_kernel<<<NSETS / 256, 256, 0, stream>>>(
        player_locs, actor_locs, flags, mask, W1, b1, Hp, scales);

    // Kernel 2: 8 waves/block * 16 rows/wave = 128 rows/block.
    se_gemm2_wmma_kernel<<<NSETS / 128, 256, 0, stream>>>(
        Hp, scales, W2, b2, out);
}
